// DTEN_38508676776404
// MI455X (gfx1250) — matrix operand order repaired and checked by reading, not compiled, not run on hardware
//
#include <hip/hip_runtime.h>
#include <hip/hip_bf16.h>

typedef __attribute__((ext_vector_type(16))) _Float16 v16h;
typedef __attribute__((ext_vector_type(8)))  _Float16 h8;
typedef __attribute__((ext_vector_type(2)))  __fp16   fp16x2;
typedef __attribute__((ext_vector_type(8)))  float    v8f;

#define NB        2
#define LQ        5376      // 4096 + 1024 + 256
#define TOK       (NB*LQ)   // 10752
#define DMODEL    256

static __host__ __device__ inline int cdiv_i(int a, int b) { return (a + b - 1) / b; }

__device__ __forceinline__ h8 pack8(float4 f0, float4 f1)
{
    union { h8 v; fp16x2 p[4]; } u;
    u.p[0] = __builtin_amdgcn_cvt_pkrtz(f0.x, f0.y);
    u.p[1] = __builtin_amdgcn_cvt_pkrtz(f0.z, f0.w);
    u.p[2] = __builtin_amdgcn_cvt_pkrtz(f1.x, f1.y);
    u.p[3] = __builtin_amdgcn_cvt_pkrtz(f1.z, f1.w);
    return u.v;
}

// ---------------------------------------------------------------------------
// WMMA GEMM, 128x64 block tile, 8 waves, each wave 32x32 (4 WMMAs / k-chunk).
//   C[m,n] = act( sum_k A[m,k]*B[k,n] + bias )
// A: [M][K]-strided, contiguous k (stride sAm).
// BROW=1: B stored [N][K] row-major (linear weights), stride sB.
// BROW=0: B stored [K][N], contiguous n, stride sB.
// K % 32 == 0 required (true for all uses). M,N tails: clamped loads,
// guarded stores.
// grid = (cdiv(N,64), cdiv(M,128), batches); block = 256.
// ---------------------------------------------------------------------------
template<int BROW>
__global__ __launch_bounds__(256)
void wmma_gemm(const float* __restrict__ A, long sAm, long sAb,
               const float* __restrict__ B, long sB,  long sBb,
               float* __restrict__ C, long sCm, long sCb,
               int M, int N, int K,
               const float* __restrict__ bias, int bias_mode, int act)
{
    __shared__ _Float16 As[128][32];
    __shared__ _Float16 Bs[64][32];    // stored transposed: [n][k]

    const int z = blockIdx.z;
    A += (long)z * sAb;  B += (long)z * sBb;  C += (long)z * sCb;

    const int m0 = blockIdx.y * 128;
    const int n0 = blockIdx.x * 64;
    const int t    = threadIdx.x;
    const int w    = t >> 5;
    const int lane = t & 31;
    const int half = lane >> 4;
    const int lm   = lane & 15;
    const int wm   = (w & 3) * 32;
    const int wn   = (w >> 2) * 32;

    // staging coords + hoisted running pointers
    const int arow = t >> 1, aseg = (t & 1) * 16;
    const int brow = t >> 2, bseg = (t & 3) * 8;
    const int bkr  = t >> 3, bnn  = (t & 7) * 8;

    const float* acur = A + (long)min(m0 + arow, M - 1) * sAm + aseg;
    const float* bcur;
    long bstep;
    if (BROW) { bcur = B + (long)min(n0 + brow, N - 1) * sB + bseg; bstep = 32; }
    else      { bcur = B + (long)bkr * sB + n0 + bnn;              bstep = 32 * sB; }

    v8f acc00 = {}, acc01 = {}, acc10 = {}, acc11 = {};

    for (int k0 = 0; k0 < K; k0 += 32) {
        {   // ---- stage A (4x b128, packed cvt) ----
            const float4* ap = (const float4*)acur;
            float4 f0 = ap[0], f1 = ap[1], f2 = ap[2], f3 = ap[3];
            *(h8*)&As[arow][aseg]     = pack8(f0, f1);
            *(h8*)&As[arow][aseg + 8] = pack8(f2, f3);
        }
        if (BROW) {
            const float4* bp = (const float4*)bcur;
            float4 f0 = bp[0], f1 = bp[1];
            *(h8*)&Bs[brow][bseg] = pack8(f0, f1);
        } else {
            const float4* bp = (const float4*)bcur;
            float4 f0 = bp[0], f1 = bp[1];
            h8 u = pack8(f0, f1);
            #pragma unroll
            for (int j = 0; j < 8; j++) Bs[bnn + j][bkr] = u[j];
        }
        acur += 32;
        bcur += bstep;
        __syncthreads();

        // ---- fragments (ISA layouts) ----
        v16h a0, a1, b0, b1;
        const _Float16* ap0 = &As[wm + lm][half * 8];
        const _Float16* ap1 = &As[wm + 16 + lm][half * 8];
        #pragma unroll
        for (int e = 0; e < 8; e++) {
            a0[e] = ap0[e]; a0[8 + e] = ap0[16 + e];
            a1[e] = ap1[e]; a1[8 + e] = ap1[16 + e];
        }
        const _Float16* bp0 = &Bs[wn + lm][half * 8];
        const _Float16* bp1 = &Bs[wn + 16 + lm][half * 8];
        #pragma unroll
        for (int e = 0; e < 8; e++) {
            b0[e] = bp0[e]; b0[8 + e] = bp0[16 + e];
            b1[e] = bp1[e]; b1[8 + e] = bp1[16 + e];
        }

        acc00 = __builtin_amdgcn_wmma_f32_16x16x32_f16(false, a0, false, b0, (short)0, acc00, false, false);
        acc01 = __builtin_amdgcn_wmma_f32_16x16x32_f16(false, a0, false, b1, (short)0, acc01, false, false);
        acc10 = __builtin_amdgcn_wmma_f32_16x16x32_f16(false, a1, false, b0, (short)0, acc10, false, false);
        acc11 = __builtin_amdgcn_wmma_f32_16x16x32_f16(false, a1, false, b1, (short)0, acc11, false, false);
        __syncthreads();
    }

    #pragma unroll
    for (int mt = 0; mt < 2; mt++) {
        #pragma unroll
        for (int r = 0; r < 8; r++) {
            int mi = m0 + wm + mt * 16 + half * 8 + r;
            if (mi >= M) continue;
            #pragma unroll
            for (int nt = 0; nt < 2; nt++) {
                int ni = n0 + wn + nt * 16 + lm;
                if (ni >= N) continue;
                float v;
                if (mt == 0) v = nt ? acc01[r] : acc00[r];
                else         v = nt ? acc11[r] : acc10[r];
                if (bias_mode == 1)      v += bias[mi];
                else if (bias_mode == 2) v += bias[ni];
                if (act == 1) v = v > 0.f ? v : 0.f;
                C[(long)mi * sCm + ni] = v;
            }
        }
    }
}

// ---------------------------------------------------------------------------
// 3x3 conv (pad=1) as implicit GEMM, 128x64 tile, branch-free border gather.
// ---------------------------------------------------------------------------
__global__ __launch_bounds__(256)
void wmma_conv3x3(const float* __restrict__ Wt,
                  const float* __restrict__ X,
                  float* __restrict__ C,
                  const float* __restrict__ bias,
                  int Cin, int H, int Wd)
{
    __shared__ _Float16 As[128][32];
    __shared__ _Float16 Bs[64][32];

    const int HW = H * Wd;
    const int M = 256, K = Cin * 9;
    const int z = blockIdx.z;
    X += (long)z * Cin * HW;
    C += (long)z * 256 * HW;

    const int m0 = blockIdx.y * 128;
    const int n0 = blockIdx.x * 64;
    const int t    = threadIdx.x;
    const int w    = t >> 5;
    const int lane = t & 31;
    const int half = lane >> 4;
    const int lm   = lane & 15;
    const int wm   = (w & 3) * 32;
    const int wn   = (w >> 2) * 32;

    const int arow = t >> 1, aseg = (t & 1) * 16;
    const int brow = t >> 2, bseg = (t & 3) * 8;
    const int by = (n0 + brow) / Wd, bx = (n0 + brow) % Wd;

    const float* acur = Wt + (long)min(m0 + arow, M - 1) * K + aseg;

    v8f acc00 = {}, acc01 = {}, acc10 = {}, acc11 = {};

    for (int k0 = 0; k0 < K; k0 += 32) {
        {
            const float4* ap = (const float4*)acur;
            float4 f0 = ap[0], f1 = ap[1], f2 = ap[2], f3 = ap[3];
            *(h8*)&As[arow][aseg]     = pack8(f0, f1);
            *(h8*)&As[arow][aseg + 8] = pack8(f2, f3);
            acur += 32;
        }
        {
            float fv[8];
            #pragma unroll
            for (int j = 0; j < 8; j++) {
                int ki = k0 + bseg + j;
                int cch = ki / 9, r = ki % 9;
                int y = by + r / 3 - 1;
                int x = bx + r % 3 - 1;
                bool inb = ((unsigned)y < (unsigned)H) & ((unsigned)x < (unsigned)Wd);
                int yc = min(max(y, 0), H - 1);
                int xc = min(max(x, 0), Wd - 1);
                float v = X[(long)cch * HW + yc * Wd + xc];   // always in-bounds
                fv[j] = inb ? v : 0.f;
            }
            union { h8 v; fp16x2 p[4]; } u;
            #pragma unroll
            for (int j = 0; j < 4; j++)
                u.p[j] = __builtin_amdgcn_cvt_pkrtz(fv[2 * j], fv[2 * j + 1]);
            *(h8*)&Bs[brow][bseg] = u.v;
        }
        __syncthreads();

        v16h a0, a1, b0, b1;
        const _Float16* ap0 = &As[wm + lm][half * 8];
        const _Float16* ap1 = &As[wm + 16 + lm][half * 8];
        #pragma unroll
        for (int e = 0; e < 8; e++) {
            a0[e] = ap0[e]; a0[8 + e] = ap0[16 + e];
            a1[e] = ap1[e]; a1[8 + e] = ap1[16 + e];
        }
        const _Float16* bp0 = &Bs[wn + lm][half * 8];
        const _Float16* bp1 = &Bs[wn + 16 + lm][half * 8];
        #pragma unroll
        for (int e = 0; e < 8; e++) {
            b0[e] = bp0[e]; b0[8 + e] = bp0[16 + e];
            b1[e] = bp1[e]; b1[8 + e] = bp1[16 + e];
        }

        acc00 = __builtin_amdgcn_wmma_f32_16x16x32_f16(false, a0, false, b0, (short)0, acc00, false, false);
        acc01 = __builtin_amdgcn_wmma_f32_16x16x32_f16(false, a0, false, b1, (short)0, acc01, false, false);
        acc10 = __builtin_amdgcn_wmma_f32_16x16x32_f16(false, a1, false, b0, (short)0, acc10, false, false);
        acc11 = __builtin_amdgcn_wmma_f32_16x16x32_f16(false, a1, false, b1, (short)0, acc11, false, false);
        __syncthreads();
    }

    #pragma unroll
    for (int mt = 0; mt < 2; mt++) {
        #pragma unroll
        for (int r = 0; r < 8; r++) {
            int mi = m0 + wm + mt * 16 + half * 8 + r;
            #pragma unroll
            for (int nt = 0; nt < 2; nt++) {
                int ni = n0 + wn + nt * 16 + lm;
                float v;
                if (mt == 0) v = nt ? acc01[r] : acc00[r];
                else         v = nt ? acc11[r] : acc10[r];
                C[(long)mi * HW + ni] = v + bias[mi];
            }
        }
    }
}

// ---------------------------------------------------------------------------
__global__ __launch_bounds__(256)
void group_norm_kernel(float* __restrict__ p, const float* __restrict__ gw,
                       const float* __restrict__ gb, int HW)
{
    __shared__ float rs[256], rs2[256];
    const int n = blockIdx.x / 32, g = blockIdx.x % 32;
    float* base = p + ((long)n * 256 + g * 8) * HW;
    const int cnt = 8 * HW;
    const int t = threadIdx.x;
    float s = 0.f, ss = 0.f;
    for (int i = t; i < cnt; i += 256) { float v = base[i]; s += v; ss += v * v; }
    rs[t] = s; rs2[t] = ss;
    __syncthreads();
    for (int o = 128; o > 0; o >>= 1) {
        if (t < o) { rs[t] += rs[t + o]; rs2[t] += rs2[t + o]; }
        __syncthreads();
    }
    float mean = rs[0] / (float)cnt;
    float var  = rs2[0] / (float)cnt - mean * mean;
    float inv  = rsqrtf(var + 1e-5f);
    for (int i = t; i < cnt; i += 256) {
        int cl = i / HW;
        base[i] = (base[i] - mean) * inv * gw[g * 8 + cl] + gb[g * 8 + cl];
    }
}

// ---------------------------------------------------------------------------
__global__ void maskflat_kernel(const unsigned char* __restrict__ pad,
                                unsigned char* __restrict__ mf)
{
    int idx = blockIdx.x * 256 + threadIdx.x;
    if (idx >= TOK) return;
    int n = idx / LQ, tq = idx % LQ;
    int H, S;
    if (tq < 4096)      { H = 64; S = 0; }
    else if (tq < 5120) { H = 32; S = 4096; }
    else                { H = 16; S = 5120; }
    int hw = tq - S;
    int i = hw / H, j = hw % H;
    int fac = 64 / H;
    mf[idx] = pad[n * 4096 + (i * fac) * 64 + (j * fac)];
}

__global__ void vr_kernel(const unsigned char* __restrict__ mf, float* __restrict__ vr)
{
    int idx = threadIdx.x;
    if (idx >= NB * 3) return;
    int n = idx / 3, l = idx % 3;
    int H = (l == 0) ? 64 : (l == 1) ? 32 : 16;
    int S = (l == 0) ? 0  : (l == 1) ? 4096 : 5120;
    const unsigned char* m = mf + n * LQ + S;
    float vh = 0.f, vw = 0.f;
    for (int i = 0; i < H; i++) vh += m[i * H] ? 0.f : 1.f;
    for (int j = 0; j < H; j++) vw += m[j] ? 0.f : 1.f;
    vr[idx * 2 + 0] = vw / (float)H;
    vr[idx * 2 + 1] = vh / (float)H;
}

// ---------------------------------------------------------------------------
__global__ __launch_bounds__(256)
void build_src_pos_kernel(const float* __restrict__ p3, const float* __restrict__ p4,
                          const float* __restrict__ p5,
                          const unsigned char* __restrict__ mf,
                          const float* __restrict__ lvl_emb,
                          float* __restrict__ srcflat, float* __restrict__ q)
{
    long idx = (long)blockIdx.x * 256 + threadIdx.x;
    if (idx >= (long)TOK * 256) return;
    int c   = (int)(idx & 255);
    int tok = (int)(idx >> 8);
    int n   = tok / LQ;
    int tq  = tok % LQ;
    int l, H, S; const float* p;
    if (tq < 4096)      { l = 0; H = 64; S = 0;    p = p3; }
    else if (tq < 5120) { l = 1; H = 32; S = 4096; p = p4; }
    else                { l = 2; H = 16; S = 5120; p = p5; }
    int hw = tq - S;
    int i = hw / H, j = hw % H;

    float sv = p[((long)n * 256 + c) * (H * H) + hw];
    srcflat[idx] = sv;

    const unsigned char* m = mf + n * LQ + S;
    const float twopi = 6.283185307179586f;
    float pe;
    if (c < 128) {
        float ys = 0.f, yd = 0.f;
        for (int ii = 0; ii < H; ii++) {
            float nm = m[ii * H + j] ? 0.f : 1.f;
            yd += nm;
            if (ii <= i) ys += nm;
        }
        float ye = ys / (yd + 1e-6f) * twopi;
        float dt = powf(10000.f, (float)(2 * (c >> 1)) / 128.f);
        float v = ye / dt;
        pe = (c & 1) ? cosf(v) : sinf(v);
    } else {
        int cc = c - 128;
        float xs = 0.f, xd = 0.f;
        for (int jj = 0; jj < H; jj++) {
            float nm = m[i * H + jj] ? 0.f : 1.f;
            xd += nm;
            if (jj <= j) xs += nm;
        }
        float xe = xs / (xd + 1e-6f) * twopi;
        float dt = powf(10000.f, (float)(2 * (cc >> 1)) / 128.f);
        float v = xe / dt;
        pe = (cc & 1) ? cosf(v) : sinf(v);
    }
    q[idx] = sv + pe + lvl_emb[l * 256 + c];
}

__global__ void mask_value_kernel(float* __restrict__ value, const unsigned char* __restrict__ mf)
{
    long idx = (long)blockIdx.x * 256 + threadIdx.x;
    if (idx >= (long)TOK * 256) return;
    if (mf[idx >> 8]) value[idx] = 0.f;
}

__global__ void softmax_aw_kernel(float* __restrict__ awr)
{
    int idx = blockIdx.x * 256 + threadIdx.x;
    if (idx >= TOK * 8) return;
    float* a = awr + (long)(idx >> 3) * 96 + (idx & 7) * 12;
    float mx = -1e30f;
    for (int k = 0; k < 12; k++) mx = fmaxf(mx, a[k]);
    float e[12], s = 0.f;
    for (int k = 0; k < 12; k++) { e[k] = expf(a[k] - mx); s += e[k]; }
    float inv = 1.f / s;
    for (int k = 0; k < 12; k++) a[k] = e[k] * inv;
}

// ---------------------------------------------------------------------------
__device__ __forceinline__ float fetch_v(const float* __restrict__ value,
                                         int n, int S, int H, int W,
                                         int y, int x, int ch)
{
    bool inb = ((unsigned)x < (unsigned)W) & ((unsigned)y < (unsigned)H);
    int yc = min(max(y, 0), H - 1), xc = min(max(x, 0), W - 1);
    float v = value[((long)(n * LQ + S + yc * W + xc)) * 256 + ch];
    return inb ? v : 0.f;
}

__global__ __launch_bounds__(256)
void deform_sample_kernel(const float* __restrict__ value,
                          const float* __restrict__ off,
                          const float* __restrict__ aw,
                          const float* __restrict__ vr,
                          float* __restrict__ out)
{
    int gid  = (blockIdx.x * 256 + threadIdx.x) >> 5;
    int lane = threadIdx.x & 31;
    if (gid >= TOK * 8) return;
    int head = gid & 7;
    int qg   = gid >> 3;
    int n    = qg / LQ;
    int tq   = qg % LQ;

    int lq, Hq, Sq;
    if (tq < 4096)      { lq = 0; Hq = 64; Sq = 0; }
    else if (tq < 5120) { lq = 1; Hq = 32; Sq = 4096; }
    else                { lq = 2; Hq = 16; Sq = 5120; }
    int hw = tq - Sq;
    float ry0 = ((float)(hw / Hq) + 0.5f) / (vr[(n * 3 + lq) * 2 + 1] * (float)Hq);
    float rx0 = ((float)(hw % Hq) + 0.5f) / (vr[(n * 3 + lq) * 2 + 0] * (float)Hq);

    const float* offp = off + (long)qg * 192 + head * 24;
    const float* awp  = aw  + (long)qg * 96  + head * 12;
    const int ch = head * 32 + lane;

    const int Hs[3] = {64, 32, 16};
    const int Ss[3] = {0, 4096, 5120};

    float acc = 0.f;
    #pragma unroll
    for (int l = 0; l < 3; l++) {
        int H = Hs[l], W = Hs[l], S = Ss[l];
        float refx = rx0 * vr[(n * 3 + l) * 2 + 0];
        float refy = ry0 * vr[(n * 3 + l) * 2 + 1];
        #pragma unroll
        for (int p = 0; p < 4; p++) {
            float ox = offp[(l * 4 + p) * 2 + 0];
            float oy = offp[(l * 4 + p) * 2 + 1];
            float wgt = awp[l * 4 + p];
            float sx = (refx + ox / (float)W) * (float)W - 0.5f;
            float sy = (refy + oy / (float)H) * (float)H - 0.5f;
            int x0 = (int)floorf(sx), y0 = (int)floorf(sy);
            float fx = sx - (float)x0, fy = sy - (float)y0;
            float g00 = fetch_v(value, n, S, H, W, y0,     x0,     ch);
            float g01 = fetch_v(value, n, S, H, W, y0,     x0 + 1, ch);
            float g10 = fetch_v(value, n, S, H, W, y0 + 1, x0,     ch);
            float g11 = fetch_v(value, n, S, H, W, y0 + 1, x0 + 1, ch);
            float top = g00 * (1.f - fx) + g01 * fx;
            float bot = g10 * (1.f - fx) + g11 * fx;
            acc += wgt * (top * (1.f - fy) + bot * fy);
        }
    }
    out[(long)qg * 256 + ch] = acc;
}

// ---------------------------------------------------------------------------
__global__ __launch_bounds__(256)
void add_ln_kernel(const float* __restrict__ x, const float* __restrict__ y,
                   const float* __restrict__ w, const float* __restrict__ b,
                   float* __restrict__ out, int ntok)
{
    int tok  = (blockIdx.x * 256 + threadIdx.x) >> 5;
    int lane = threadIdx.x & 31;
    if (tok >= ntok) return;
    const float4* xp = (const float4*)(x + (long)tok * 256) + lane * 2;
    const float4* yp = (const float4*)(y + (long)tok * 256) + lane * 2;
    float4 xa = xp[0], xb = xp[1], ya = yp[0], yb = yp[1];
    float v[8];
    v[0]=xa.x+ya.x; v[1]=xa.y+ya.y; v[2]=xa.z+ya.z; v[3]=xa.w+ya.w;
    v[4]=xb.x+yb.x; v[5]=xb.y+yb.y; v[6]=xb.z+yb.z; v[7]=xb.w+yb.w;
    float s = 0.f;
    #pragma unroll
    for (int j = 0; j < 8; j++) s += v[j];
    #pragma unroll
    for (int o = 16; o > 0; o >>= 1) s += __shfl_xor(s, o, 32);
    float mu = s / 256.f;
    float ss = 0.f;
    #pragma unroll
    for (int j = 0; j < 8; j++) { float d = v[j] - mu; ss += d * d; }
    #pragma unroll
    for (int o = 16; o > 0; o >>= 1) ss += __shfl_xor(ss, o, 32);
    float inv = rsqrtf(ss / 256.f + 1e-5f);
    float* op = out + (long)tok * 256;
    #pragma unroll
    for (int j = 0; j < 8; j++)
        op[lane * 8 + j] = (v[j] - mu) * inv * w[lane * 8 + j] + b[lane * 8 + j];
}

__global__ void extract_o_kernel(const float* __restrict__ src2, float* __restrict__ o)
{
    int idx = blockIdx.x * 256 + threadIdx.x;
    if (idx >= NB * 256 * 256) return;
    int n  = idx / (256 * 256);
    int rm = idx % (256 * 256);
    int c  = rm / 256;
    int hw = rm % 256;
    o[idx] = src2[((long)(n * LQ + 5120 + hw)) * 256 + c];
}

__global__ __launch_bounds__(256)
void resize_add_prelu_kernel(const float* __restrict__ in, int Hi, int Wi,
                             const float* __restrict__ conv,
                             const float* __restrict__ alpha_p,
                             float* __restrict__ out, int Ho, int Wo)
{
    long total = (long)NB * 256 * Ho * Wo;
    long idx = (long)blockIdx.x * 256 + threadIdx.x;
    if (idx >= total) return;
    int hw = (int)(idx % (Ho * Wo));
    long nc = idx / (Ho * Wo);
    int yy = hw / Wo, xx = hw % Wo;
    float sy = ((float)yy + 0.5f) * (float)Hi / (float)Ho - 0.5f;
    float sx = ((float)xx + 0.5f) * (float)Wi / (float)Wo - 0.5f;
    int y0f = (int)floorf(sy), x0f = (int)floorf(sx);
    float fy = sy - (float)y0f, fx = sx - (float)x0f;
    int y0 = min(max(y0f, 0), Hi - 1), y1 = min(max(y0f + 1, 0), Hi - 1);
    int x0 = min(max(x0f, 0), Wi - 1), x1 = min(max(x0f + 1, 0), Wi - 1);
    const float* ip = in + nc * (long)(Hi * Wi);
    float v = (ip[y0 * Wi + x0] * (1.f - fx) + ip[y0 * Wi + x1] * fx) * (1.f - fy) +
              (ip[y1 * Wi + x0] * (1.f - fx) + ip[y1 * Wi + x1] * fx) * fy;
    float cv = conv[idx];
    float a = alpha_p[0];
    out[idx] = v + (cv >= 0.f ? cv : a * cv);
}

// ---------------------------------------------------------------------------
extern "C" void kernel_launch(void* const* d_in, const int* in_sizes, int n_in,
                              void* d_out, int out_size, void* d_ws, size_t ws_size,
                              hipStream_t stream)
{
    const float* x0   = (const float*)d_in[0];
    const float* x1   = (const float*)d_in[1];
    const float* x2   = (const float*)d_in[2];
    const unsigned char* pad = (const unsigned char*)d_in[3];
    const float* p3_w = (const float*)d_in[4];  const float* p3_b = (const float*)d_in[5];
    const float* p4_w = (const float*)d_in[6];  const float* p4_b = (const float*)d_in[7];
    const float* p5_w = (const float*)d_in[8];  const float* p5_b = (const float*)d_in[9];
    const float* gn3_w = (const float*)d_in[10]; const float* gn3_b = (const float*)d_in[11];
    const float* gn4_w = (const float*)d_in[12]; const float* gn4_b = (const float*)d_in[13];
    const float* gn5_w = (const float*)d_in[14]; const float* gn5_b = (const float*)d_in[15];
    const float* lvl_emb = (const float*)d_in[16];
    const float* so_w = (const float*)d_in[17]; const float* so_b = (const float*)d_in[18];
    const float* aw_w = (const float*)d_in[19]; const float* aw_b = (const float*)d_in[20];
    const float* vp_w = (const float*)d_in[21]; const float* vp_b = (const float*)d_in[22];
    const float* op_w = (const float*)d_in[23]; const float* op_b = (const float*)d_in[24];
    const float* ln1_w = (const float*)d_in[25]; const float* ln1_b = (const float*)d_in[26];
    const float* ffn1_w = (const float*)d_in[27]; const float* ffn1_b = (const float*)d_in[28];
    const float* ffn2_w = (const float*)d_in[29]; const float* ffn2_b = (const float*)d_in[30];
    const float* ln2_w = (const float*)d_in[31]; const float* ln2_b = (const float*)d_in[32];
    const float* d3_w = (const float*)d_in[33]; const float* d3_b = (const float*)d_in[34]; const float* d3_a = (const float*)d_in[35];
    const float* d4_w = (const float*)d_in[36]; const float* d4_b = (const float*)d_in[37]; const float* d4_a = (const float*)d_in[38];
    const float* d5_w = (const float*)d_in[39]; const float* d5_b = (const float*)d_in[40]; const float* d5_a = (const float*)d_in[41];

    // -------- workspace arena (floats) with reuse --------
    float* ws = (float*)d_ws;
    size_t off_ = 0;
    auto alloc = [&](size_t nf) { float* p = ws + off_; off_ += nf; return p; };
    float* srcflat = alloc((size_t)TOK * 256);
    float* qbuf    = alloc((size_t)TOK * 256);          // q; reused as src1
    float* value   = alloc((size_t)TOK * 256);          // reused as ffn2 out
    float* offb    = alloc((size_t)TOK * 192);
    float* awr     = alloc((size_t)TOK * 96);           // reused as o16
    float* samp    = alloc((size_t)TOK * 256);          // reused as src2
    float* attn    = alloc((size_t)TOK * 256);          // reused as c3
    float* hbuf    = alloc((size_t)TOK * 1024);         // p3/p4/p5 early; ffn hidden; o64 late
    float* c45     = alloc((size_t)(524288 + 131072 + 524288));
    float* vrbuf   = alloc(16);
    unsigned char* mf = (unsigned char*)alloc(4096);

    float* p3 = hbuf;
    float* p4 = hbuf + (size_t)NB * 256 * 4096;
    float* p5 = p4 + (size_t)NB * 256 * 1024;
    float* src1 = qbuf;
    float* ffo  = value;
    float* src2 = samp;
    float* c3   = attn;
    float* o16  = awr;
    float* o64  = hbuf;
    float* c4   = c45;
    float* c5   = c45 + 524288;
    float* o32  = c45 + 655360;

    // ---- 1x1 convs (2048->256): C[o,hw] = W[o,c] * X[c,hw]  (B = [K][N]) ----
    wmma_gemm<0><<<dim3(64, 2, NB), 256, 0, stream>>>(
        p3_w, 2048, 0,   x0, 4096, (long)2048 * 4096,
        p3, 4096, (long)256 * 4096, 256, 4096, 2048, p3_b, 1, 0);
    wmma_gemm<0><<<dim3(16, 2, NB), 256, 0, stream>>>(
        p4_w, 2048, 0,   x1, 1024, (long)2048 * 1024,
        p4, 1024, (long)256 * 1024, 256, 1024, 2048, p4_b, 1, 0);
    wmma_gemm<0><<<dim3(4, 2, NB), 256, 0, stream>>>(
        p5_w, 2048, 0,   x2, 256, (long)2048 * 256,
        p5, 256, (long)256 * 256, 256, 256, 2048, p5_b, 1, 0);

    // ---- GroupNorm ----
    group_norm_kernel<<<NB * 32, 256, 0, stream>>>(p3, gn3_w, gn3_b, 4096);
    group_norm_kernel<<<NB * 32, 256, 0, stream>>>(p4, gn4_w, gn4_b, 1024);
    group_norm_kernel<<<NB * 32, 256, 0, stream>>>(p5, gn5_w, gn5_b, 256);

    // ---- masks / valid ratios / tokens + pos ----
    maskflat_kernel<<<cdiv_i(TOK, 256), 256, 0, stream>>>(pad, mf);
    vr_kernel<<<1, 32, 0, stream>>>(mf, vrbuf);
    build_src_pos_kernel<<<cdiv_i(TOK * 256, 256), 256, 0, stream>>>(
        p3, p4, p5, mf, lvl_emb, srcflat, qbuf);

    // ---- MSDeformAttn projections (B = weights [N][K]) ----
    const int GY = cdiv_i(TOK, 128);   // 84
    wmma_gemm<1><<<dim3(4, GY, 1), 256, 0, stream>>>(
        srcflat, 256, 0,  vp_w, 256, 0,
        value, 256, 0, TOK, 256, 256, vp_b, 2, 0);
    mask_value_kernel<<<cdiv_i(TOK * 256, 256), 256, 0, stream>>>(value, mf);
    wmma_gemm<1><<<dim3(3, GY, 1), 256, 0, stream>>>(
        qbuf, 256, 0,  so_w, 256, 0,
        offb, 192, 0, TOK, 192, 256, so_b, 2, 0);
    wmma_gemm<1><<<dim3(2, GY, 1), 256, 0, stream>>>(
        qbuf, 256, 0,  aw_w, 256, 0,
        awr, 96, 0, TOK, 96, 256, aw_b, 2, 0);
    softmax_aw_kernel<<<cdiv_i(TOK * 8, 256), 256, 0, stream>>>(awr);

    // ---- deformable sampling + output projection ----
    deform_sample_kernel<<<TOK, 256, 0, stream>>>(value, offb, awr, vrbuf, samp);
    wmma_gemm<1><<<dim3(4, GY, 1), 256, 0, stream>>>(
        samp, 256, 0,  op_w, 256, 0,
        attn, 256, 0, TOK, 256, 256, op_b, 2, 0);

    // ---- residual + LN1 ----
    add_ln_kernel<<<cdiv_i(TOK * 32, 256), 256, 0, stream>>>(
        srcflat, attn, ln1_w, ln1_b, src1, TOK);

    // ---- FFN ----
    wmma_gemm<1><<<dim3(16, GY, 1), 256, 0, stream>>>(
        src1, 256, 0,  ffn1_w, 256, 0,
        hbuf, 1024, 0, TOK, 1024, 256, ffn1_b, 2, 1);
    wmma_gemm<1><<<dim3(4, GY, 1), 256, 0, stream>>>(
        hbuf, 1024, 0,  ffn2_w, 1024, 0,
        ffo, 256, 0, TOK, 256, 1024, ffn2_b, 2, 0);
    add_ln_kernel<<<cdiv_i(TOK * 32, 256), 256, 0, stream>>>(
        src1, ffo, ln2_w, ln2_b, src2, TOK);

    // ---- 3x3 conv branches (implicit GEMM, K = 2048*9) ----
    wmma_conv3x3<<<dim3(64, 2, NB), 256, 0, stream>>>(d3_w, x0, c3, d3_b, 2048, 64, 64);
    wmma_conv3x3<<<dim3(16, 2, NB), 256, 0, stream>>>(d4_w, x1, c4, d4_b, 2048, 32, 32);
    wmma_conv3x3<<<dim3(4, 2, NB), 256, 0, stream>>>(d5_w, x2, c5, d5_b, 2048, 16, 16);

    // ---- decoder: extract level-2 tokens, cascade resize + prelu-add ----
    extract_o_kernel<<<cdiv_i(NB * 256 * 256, 256), 256, 0, stream>>>(src2, o16);
    resize_add_prelu_kernel<<<cdiv_i(NB * 256 * 4096, 256), 256, 0, stream>>>(
        o16, 16, 16, c3, d3_a, o64, 64, 64);
    resize_add_prelu_kernel<<<cdiv_i(NB * 256 * 1024, 256), 256, 0, stream>>>(
        o64, 64, 64, c4, d4_a, o32, 32, 32);
    resize_add_prelu_kernel<<<cdiv_i(NB * 256 * 256, 256), 256, 0, stream>>>(
        o32, 32, 32, c5, d5_a, (float*)d_out, 16, 16);

    (void)in_sizes; (void)n_in; (void)out_size; (void)ws_size;
}
